// OnnxAttForward_17489106829848
// MI455X (gfx1250) — matrix operand order repaired and checked by reading, not compiled, not run on hardware
//
#include <hip/hip_runtime.h>
#include <hip/hip_bf16.h>
#include <math.h>

#define T 32768
#define DUNITS 1024
#define ATT_DIM 512
#define EPROJS 512
#define CCH 32
#define KF 15   // 31-tap conv

typedef __attribute__((ext_vector_type(16))) _Float16 v16h;
typedef __attribute__((ext_vector_type(8)))  float    v8f;

// Branch-free tanh: prefer gfx1250 V_TANH_F32, else exp2/rcp ladder (no EXEC churn)
__device__ __forceinline__ float fast_tanh(float x) {
#if __has_builtin(__builtin_amdgcn_tanhf)
  return __builtin_amdgcn_tanhf(x);
#elif __has_builtin(__builtin_amdgcn_tanh_f32)
  return __builtin_amdgcn_tanh_f32(x);
#else
  float ax = __builtin_fabsf(x);
  float u  = __builtin_amdgcn_exp2f(ax * -2.885390082f);  // exp(-2|x|)
  float t  = (1.0f - u) * __builtin_amdgcn_rcpf(1.0f + u);
  return __builtin_copysignf(t, x);
#endif
}

// exp(x) for x <= 0 (post max-shift): single v_exp_f32, branch-free
__device__ __forceinline__ float fast_exp_nonpos(float x) {
  return __builtin_amdgcn_exp2f(x * 1.4426950408889634f);
}

// ---------------- K0: dec_proj[a] = sum_k W_dec[a,k] * dec_z[k] ----------------
__global__ void __launch_bounds__(256)
k0_decproj(const float* __restrict__ W_dec,
           const float* __restrict__ dec_z,
           float* __restrict__ dec_proj) {
  int a = blockIdx.x * blockDim.x + threadIdx.x;
  if (a >= ATT_DIM) return;
  const float* row = W_dec + (size_t)a * DUNITS;
  float s = 0.f;
  #pragma unroll 8
  for (int k = 0; k < DUNITS; ++k) s += row[k] * dec_z[k];
  dec_proj[a] = s;
}

// ---------------- K1: fused conv -> WMMA GEMM -> tanh energy ----------------
// Each wave handles a 16-row t-tile. A (16x32 f16) built in contiguous k order
// (k = j + 16*half): lane l holds row l%16; half-wave selects K-chunk.
__global__ void __launch_bounds__(256)
k1_energy(const float* __restrict__ att_prev,
          const float* __restrict__ pre,      // [T,512]
          const float* __restrict__ conv_w,   // [32,31]
          const float* __restrict__ conv_b,   // [32]
          const float* __restrict__ W_att,    // [512,32]
          const float* __restrict__ W_g,      // [512]
          const float* __restrict__ b_g,      // [1]
          const float* __restrict__ mask,     // [T]
          const float* __restrict__ dec_proj, // [512]
          float* __restrict__ e_out)          // [T] scaled energies
{
  const int tid  = threadIdx.x;
  const int wave = tid >> 5;
  const int lane = tid & 31;
  const int half = lane >> 4;     // which half-wave
  const int n    = lane & 15;     // row index within A tile / col within B,C
  const int t0   = blockIdx.x * 128 + wave * 16;
  const int row  = t0 + n;        // conv row this lane produces

  // conv window for this row (zero-padded)
  float win[31];
  #pragma unroll
  for (int u = 0; u < 31; ++u) {
    int idx = row + u - KF;
    win[u] = (idx >= 0 && idx < T) ? att_prev[idx] : 0.f;
  }

  // Build A tile: element j of v16h maps to channel K = j + 16*half
  v16h A;
  #pragma unroll
  for (int j = 0; j < 16; ++j) {
    const int c = j + (half << 4);
    const float* wrow = conv_w + c * 31;
    float acc = conv_b[c];
    #pragma unroll
    for (int u = 0; u < 31; ++u) acc += win[u] * wrow[u];
    A[j] = (_Float16)acc;
  }

  float epart[8];
  #pragma unroll
  for (int i = 0; i < 8; ++i) epart[i] = 0.f;

  #pragma unroll 2
  for (int ab = 0; ab < 32; ++ab) {
    const int a0 = ab * 16;
    // B tile (32x16 f16): lane supplies column N=n, K = j + 16*half
    const float* bsrc = W_att + (size_t)(a0 + n) * CCH + half * 16;
    v16h B;
    #pragma unroll
    for (int j = 0; j < 16; ++j) B[j] = (_Float16)bsrc[j];

    v8f C = {};
    C = __builtin_amdgcn_wmma_f32_16x16x32_f16(false, A, false, B,
                                               (short)0, C, false, false);

    const float d = dec_proj[a0 + n];
    const float g = W_g[a0 + n];
    // C layout: VGPR i -> M = i + 8*half for this lane, N = n
    #pragma unroll
    for (int i = 0; i < 8; ++i) {
      const int r = t0 + i + half * 8;
      float z = C[i] + pre[(size_t)r * ATT_DIM + a0 + n] + d;
      epart[i] += g * fast_tanh(z);
    }
  }

  // reduce over the 16 column-lanes of each half-wave
  #pragma unroll
  for (int i = 0; i < 8; ++i) {
    float v = epart[i];
    v += __shfl_xor(v, 1);
    v += __shfl_xor(v, 2);
    v += __shfl_xor(v, 4);
    v += __shfl_xor(v, 8);
    epart[i] = v;
  }
  if (n == 0) {
    const float bg = b_g[0];
    #pragma unroll
    for (int i = 0; i < 8; ++i) {
      const int r = t0 + i + half * 8;
      e_out[r] = 2.0f * (epart[i] + bg + mask[r]);   // softmax scaling folded in
    }
  }
}

// ---------------- K2: softmax + forward recurrence + renormalize ----------------
__global__ void __launch_bounds__(1024)
k2_softmax(const float* __restrict__ att_prev,
           float* __restrict__ e_buf,   // in: scaled e; out: clipped q
           float* __restrict__ out)     // d_out: [0,512)=c (zeroed), [512,..)=w
{
  __shared__ float red[1024];
  __shared__ float s_m, s_S, s_Z;
  const int tid = threadIdx.x;

  float m = -3.4e38f;
  for (int t = tid; t < T; t += 1024) m = fmaxf(m, e_buf[t]);
  red[tid] = m; __syncthreads();
  for (int s = 512; s > 0; s >>= 1) {
    if (tid < s) red[tid] = fmaxf(red[tid], red[tid + s]);
    __syncthreads();
  }
  if (tid == 0) s_m = red[0];
  __syncthreads();
  m = s_m;

  float S = 0.f;
  for (int t = tid; t < T; t += 1024) S += fast_exp_nonpos(e_buf[t] - m);
  red[tid] = S; __syncthreads();
  for (int s = 512; s > 0; s >>= 1) {
    if (tid < s) red[tid] += red[tid + s];
    __syncthreads();
  }
  if (tid == 0) s_S = red[0];
  __syncthreads();
  const float invS = 1.f / s_S;

  float Z = 0.f;
  for (int t = tid; t < T; t += 1024) {
    float p = att_prev[t] + (t > 0 ? att_prev[t - 1] : 0.f);
    float q = p * fast_exp_nonpos(e_buf[t] - m) * invS;
    q = fmaxf(q, 1e-6f);
    e_buf[t] = q;
    Z += q;
  }
  red[tid] = Z; __syncthreads();
  for (int s = 512; s > 0; s >>= 1) {
    if (tid < s) red[tid] += red[tid + s];
    __syncthreads();
  }
  if (tid == 0) s_Z = red[0];
  __syncthreads();
  const float invZ = 1.f / s_Z;

  for (int t = tid; t < T; t += 1024) out[512 + t] = e_buf[t] * invZ;
  if (tid < 512) out[tid] = 0.f;   // zero context accumulator for K3
}

// ---------------- K3: c[a] = sum_t enc_h[t,a] * w[t] ----------------
__global__ void __launch_bounds__(256)
k3_context(const float* __restrict__ enc_h,
           const float* __restrict__ w,     // final weights
           float* __restrict__ c)           // d_out[0..512)
{
  const int tid   = threadIdx.x;
  const int tbase = blockIdx.x * 64;
  float acc0 = 0.f, acc1 = 0.f;
  for (int tt = 0; tt < 64; ++tt) {
    const int t = tbase + tt;
    const float wv = w[t];
    const float* rowp = enc_h + (size_t)t * EPROJS;
    acc0 += rowp[tid] * wv;
    acc1 += rowp[tid + 256] * wv;
  }
  atomicAdd(&c[tid], acc0);
  atomicAdd(&c[tid + 256], acc1);
}

extern "C" void kernel_launch(void* const* d_in, const int* in_sizes, int n_in,
                              void* d_out, int out_size, void* d_ws, size_t ws_size,
                              hipStream_t stream) {
  const float* dec_z    = (const float*)d_in[0];
  const float* att_prev = (const float*)d_in[1];
  const float* pre      = (const float*)d_in[2];
  const float* enc_h    = (const float*)d_in[3];
  const float* mask     = (const float*)d_in[4];
  const float* conv_w   = (const float*)d_in[5];
  const float* conv_b   = (const float*)d_in[6];
  const float* W_att    = (const float*)d_in[7];
  const float* W_dec    = (const float*)d_in[8];
  const float* W_g      = (const float*)d_in[9];
  const float* b_g      = (const float*)d_in[10];
  float* out = (float*)d_out;

  float* dec_proj = (float*)d_ws;          // 512 f32
  float* e_buf    = (float*)d_ws + 1024;   // T f32

  k0_decproj<<<2, 256, 0, stream>>>(W_dec, dec_z, dec_proj);
  k1_energy<<<T / 128, 256, 0, stream>>>(att_prev, pre, conv_w, conv_b, W_att,
                                         W_g, b_g, mask, dec_proj, e_buf);
  k2_softmax<<<1, 1024, 0, stream>>>(att_prev, e_buf, out);
  k3_context<<<T / 64, 256, 0, stream>>>(enc_h, out + 512, out);
}
